// ae_tamrl_18580028522989
// MI455X (gfx1250) — compile-verified, matrix operand order checked by reading, not run on hardware
//
#include <hip/hip_runtime.h>

// ---------------------------------------------------------------------------
// Sizes (from reference)
// ---------------------------------------------------------------------------
#define B_   512
#define T_   365
#define CIN_ 32
#define H_   512
#define G4_  2048      // 4*H
#define CODE_ 128
#define OUTC_ 16
#define ROWS_ (T_ * B_)        // 186880
#define BH_   (B_ * H_)        // 262144
#define NPART_ 730             // ROWS_/256

typedef __attribute__((ext_vector_type(16))) __bf16 bf16x16;
typedef __attribute__((ext_vector_type(8)))  float  f32x8;

union FragU {
  bf16x16 v;
  uint4   q[2];
  unsigned short h[16];
};

// f32 -> bf16 (RNE) via hardware convert; f32 <- bf16 via shift.
__device__ __forceinline__ unsigned short f2bf(float f) {
  __bf16 b = (__bf16)f;
  return __builtin_bit_cast(unsigned short, b);
}
__device__ __forceinline__ float bf2f(unsigned short h) {
  return __uint_as_float(((unsigned)h) << 16);
}
__device__ __forceinline__ float sigm(float x) { return 1.0f / (1.0f + __expf(-x)); }
__device__ __forceinline__ float tanh_(float x) {
  float e = __expf(2.0f * x);
  return 1.0f - 2.0f / (e + 1.0f);     // -> -1 / +1 at extremes, no NaN
}

// ---------------------------------------------------------------------------
// CDNA5 async global->LDS copy (ASYNCcnt path).  lds_dst is a generic pointer
// into LDS: flat->LDS aperture mapping keeps the LDS byte offset in addr[31:0].
// ---------------------------------------------------------------------------
__device__ __forceinline__ void async_g2l_b128(void* lds_dst, const void* gsrc) {
  unsigned ldsoff = (unsigned)(size_t)lds_dst;
  asm volatile("global_load_async_to_lds_b128 %0, %1, off"
               :: "v"(ldsoff), "v"(gsrc) : "memory");
}
__device__ __forceinline__ void wait_async0() {
  asm volatile("s_wait_asynccnt 0" ::: "memory");
}

// Load one 16x32 bf16 fragment in the ISA's A-operand lane layout.
// lane<16:  K = kk+{0..7} and kk+{16..23};  lane>=16: K = kk+{8..15},{24..31}.
__device__ __forceinline__ bf16x16 load_frag(const unsigned short* base, int ldk,
                                             int lane, int kk) {
  int r    = lane & 15;
  int koff = (lane & 16) ? 8 : 0;
  const unsigned short* p = base + (size_t)r * ldk + (size_t)(kk + koff);
  FragU f;
  f.q[0] = *(const uint4*)(p);
  f.q[1] = *(const uint4*)(p + 16);
  return f.v;
}

// Device-scope grid barrier (counter + generation), agent scope -> L2.
__device__ __forceinline__ void grid_bar(unsigned* cnt, unsigned* gen, unsigned nb) {
  __threadfence();
  __syncthreads();
  if (threadIdx.x == 0u) {
    unsigned g = __hip_atomic_load(gen, __ATOMIC_RELAXED, __HIP_MEMORY_SCOPE_AGENT);
    unsigned arrived =
        __hip_atomic_fetch_add(cnt, 1u, __ATOMIC_ACQ_REL, __HIP_MEMORY_SCOPE_AGENT);
    if (arrived == nb - 1u) {
      __hip_atomic_store(cnt, 0u, __ATOMIC_RELAXED, __HIP_MEMORY_SCOPE_AGENT);
      __hip_atomic_fetch_add(gen, 1u, __ATOMIC_RELEASE, __HIP_MEMORY_SCOPE_AGENT);
    } else {
      while (__hip_atomic_load(gen, __ATOMIC_ACQUIRE, __HIP_MEMORY_SCOPE_AGENT) == g) {
        __builtin_amdgcn_s_sleep(2);
      }
    }
  }
  __syncthreads();
}

// ---------------------------------------------------------------------------
// 1) Weight prep: f32 -> bf16, build fused [Wih|Whh] matrices, zero barrier.
// ---------------------------------------------------------------------------
__global__ __launch_bounds__(256) void prep_kernel(
    const float* __restrict__ Wih_f, const float* __restrict__ Whh_f,
    const float* __restrict__ Wih_b, const float* __restrict__ Whh_b,
    const float* __restrict__ Wih_d, const float* __restrict__ Whh_d,
    const float* __restrict__ We,    const float* __restrict__ Wo,
    unsigned short* Wcat_f, unsigned short* Wcat_b,
    unsigned short* Wdec0,  unsigned short* Wdec1,
    unsigned short* We_bf,  unsigned short* Wo_bf, unsigned* bar) {
  size_t i = (size_t)blockIdx.x * blockDim.x + threadIdx.x;
  size_t stride = (size_t)gridDim.x * blockDim.x;
  if (i < 8) bar[i] = 0u;
  const size_t NC = (size_t)G4_ * 1024;
  for (size_t j = i; j < NC; j += stride) {
    size_t r = j >> 10, k = j & 1023;
    float vf = (k < H_) ? Wih_f[r * H_ + k] : Whh_f[r * H_ + (k - H_)];
    float vb = (k < H_) ? Wih_b[r * H_ + k] : Whh_b[r * H_ + (k - H_)];
    Wcat_f[j] = f2bf(vf);
    Wcat_b[j] = f2bf(vb);
  }
  const size_t ND = (size_t)G4_ * H_;
  for (size_t j = i; j < ND; j += stride) {
    Wdec0[j] = f2bf(Whh_d[j]);
    Wdec1[j] = f2bf(Wih_d[j] + Whh_d[j]);   // decoder t>=1: xin == h
  }
  for (size_t j = i; j < (size_t)H_ * CIN_; j += stride) We_bf[j] = f2bf(We[j]);
  for (size_t j = i; j < (size_t)CIN_ * H_; j += stride) Wo_bf[j] = f2bf(Wo[j]);
}

// ---------------------------------------------------------------------------
// 2) Encoder GEMM: h_pre[t*B+b][n] = sum_k x[b][t][k] * We[n][k]  (K=32 -> 1 WMMA)
//    `be` omitted: it cancels exactly in BatchNorm's (h - mean).
// ---------------------------------------------------------------------------
__global__ __launch_bounds__(256) void enc_gemm_kernel(
    const float* __restrict__ x, const unsigned short* __restrict__ We_bf,
    unsigned short* __restrict__ h_enc) {
  int lane = threadIdx.x & 31, wave = threadIdx.x >> 5;
  int m0 = blockIdx.x * 16;
  int n0 = (blockIdx.y * 8 + wave) * 16;
  int r = lane & 15;
  int k0 = (lane & 16) ? 8 : 0;
  int m = m0 + r, t = m >> 9, b = m & 511;
  const float* xr = x + ((size_t)b * T_ + t) * CIN_;
  FragU a;
#pragma unroll
  for (int i = 0; i < 8; i++) a.h[i] = f2bf(xr[k0 + i]);
#pragma unroll
  for (int i = 0; i < 8; i++) a.h[8 + i] = f2bf(xr[k0 + 16 + i]);
  bf16x16 bf = load_frag(We_bf + (size_t)n0 * CIN_, CIN_, lane, 0);
  f32x8 acc = {};
  acc = __builtin_amdgcn_wmma_f32_16x16x32_bf16(false, a.v, false, bf, (short)0, acc,
                                                false, false);
  int col = n0 + (lane & 15);
  int rowbase = m0 + ((lane & 16) ? 8 : 0);
#pragma unroll
  for (int rr = 0; rr < 8; rr++)
    h_enc[(size_t)(rowbase + rr) * H_ + col] = f2bf(acc[rr]);
}

// ---------------------------------------------------------------------------
// 3) BN stats, deterministic two-stage (fixed-order) reduction.
// ---------------------------------------------------------------------------
__global__ __launch_bounds__(256) void bn_part_kernel(
    const unsigned short* __restrict__ h_enc, float* __restrict__ part) {
  int p = blockIdx.x;            // 0..729, 256 rows each
  int c = threadIdx.x;
  const unsigned short* base = h_enc + (size_t)p * 256 * H_;
  float s0 = 0.f, s1 = 0.f, q0 = 0.f, q1 = 0.f;
  for (int r = 0; r < 256; r++) {
    float v0 = bf2f(base[(size_t)r * H_ + c]);
    float v1 = bf2f(base[(size_t)r * H_ + c + 256]);
    s0 += v0; q0 += v0 * v0;
    s1 += v1; q1 += v1 * v1;
  }
  part[(size_t)p * H_ + c] = s0;
  part[(size_t)p * H_ + c + 256] = s1;
  part[(size_t)NPART_ * H_ + (size_t)p * H_ + c] = q0;
  part[(size_t)NPART_ * H_ + (size_t)p * H_ + c + 256] = q1;
}

__global__ __launch_bounds__(256) void bn_finish_kernel(
    const float* __restrict__ part, float* __restrict__ stats) {
  int c = blockIdx.x * 256 + threadIdx.x;
  if (c >= H_) return;
  float s = 0.f, q = 0.f;
  for (int p = 0; p < NPART_; p++) {
    s += part[(size_t)p * H_ + c];
    q += part[(size_t)NPART_ * H_ + (size_t)p * H_ + c];
  }
  const float n = (float)ROWS_;
  float mu = s / n;
  float var = q / n - mu * mu;   // biased, matches jnp.var
  stats[c] = mu;
  stats[H_ + c] = rsqrtf(var + 1e-5f);
}

__global__ __launch_bounds__(256) void bn_apply_kernel(
    unsigned short* __restrict__ h_enc, const float* __restrict__ stats,
    const float* __restrict__ gamma, const float* __restrict__ beta) {
  size_t i = (size_t)blockIdx.x * 256 + threadIdx.x;
  int c = (int)(i & (H_ - 1));
  float v = bf2f(h_enc[i]);
  v = (v - stats[c]) * stats[H_ + c] * gamma[c] + beta[c];
  v = (v >= 0.f) ? v : 0.2f * v;
  h_enc[i] = f2bf(v);
}

// ---------------------------------------------------------------------------
// 4) Encoder LSTM (persistent, one launch per direction).
//    LDS: 4 gates x 16 rows x (1024+8) weights  +  128 x (512+8) x-tile stage.
//    Per step: async-stage x(t) into LDS, compute recurrent half from L2,
//    s_wait_asynccnt, finish x half from LDS, pointwise LSTM in registers.
// ---------------------------------------------------------------------------
#define LDW_E (1024 + 8)
#define LDX_  (512 + 8)

__global__ __launch_bounds__(256) void lstm_enc_kernel(
    const unsigned short* __restrict__ h_enc, const unsigned short* __restrict__ Wcat,
    const float* __restrict__ bias, unsigned short* hp0, unsigned short* hp1,
    float* __restrict__ hfin, unsigned* bar, int dir) {
  extern __shared__ unsigned short lds[];
  unsigned short* wlds = lds;                       // 4*16*LDW_E halves
  unsigned short* xs   = lds + 4 * 16 * LDW_E;      // 128*LDX_ halves
  int lane = threadIdx.x & 31, wave = threadIdx.x >> 5;
  int n0 = blockIdx.x * 16;
  int mblock = blockIdx.y * 128;
  int m0 = mblock + wave * 16;

  { // one-shot async weight preload: 64 rows x 1024 halves
    int row = threadIdx.x >> 2;                     // 0..63
    int part = threadIdx.x & 3;
    int g = row >> 4, rr = row & 15;
    const unsigned short* src =
        Wcat + ((size_t)(g * H_ + n0 + rr)) * 1024 + part * 256;
    unsigned short* dst = wlds + (size_t)row * LDW_E + part * 256;
#pragma unroll
    for (int i = 0; i < 256; i += 8)
      async_g2l_b128(dst + i, src + i);
  }
  wait_async0();
  __syncthreads();

  int col = n0 + (lane & 15);
  float b_i = bias[0 * H_ + col], b_f = bias[1 * H_ + col];
  float b_g = bias[2 * H_ + col], b_o = bias[3 * H_ + col];
  float cst[8];
#pragma unroll
  for (int rr = 0; rr < 8; rr++) cst[rr] = 0.f;

  unsigned short* hcur = hp0;
  unsigned short* hnxt = hp1;
  const unsigned nb = gridDim.x * gridDim.y;
  int rowbase = m0 + ((lane & 16) ? 8 : 0);
  int srow = threadIdx.x >> 1;                      // staging row 0..127
  int spart = threadIdx.x & 1;

  for (int t = 0; t < T_; t++) {
    int tsrc = dir ? (T_ - 1 - t) : t;
    // --- async stage of x(t): 128 rows x 512 halves ---
    {
      const unsigned short* src =
          h_enc + (size_t)tsrc * BH_ + (size_t)(mblock + srow) * H_ + spart * 256;
      unsigned short* dst = xs + (size_t)srow * LDX_ + spart * 256;
#pragma unroll
      for (int i = 0; i < 256; i += 8)
        async_g2l_b128(dst + i, src + i);
    }
    // --- prefetch next step's x into L2 while we compute ---
    if (t + 1 < T_) {
      int tn = dir ? (T_ - 2 - t) : (t + 1);
      __builtin_prefetch(h_enc + (size_t)tn * BH_ + (size_t)(mblock + srow) * H_ +
                             spart * 256, 0, 1);
    }
    f32x8 a0 = {}, a1 = {}, a2 = {}, a3 = {};
    // --- recurrent half (K = 512..1023) from global h (L2-resident) ---
    if (t > 0) {
      const unsigned short* hbase = hcur + (size_t)m0 * H_;
      for (int kk = 0; kk < H_; kk += 32) {
        bf16x16 af = load_frag(hbase, H_, lane, kk);
        bf16x16 w0 = load_frag(wlds + 0 * 16 * LDW_E, LDW_E, lane, kk + H_);
        bf16x16 w1 = load_frag(wlds + 1 * 16 * LDW_E, LDW_E, lane, kk + H_);
        bf16x16 w2 = load_frag(wlds + 2 * 16 * LDW_E, LDW_E, lane, kk + H_);
        bf16x16 w3 = load_frag(wlds + 3 * 16 * LDW_E, LDW_E, lane, kk + H_);
        a0 = __builtin_amdgcn_wmma_f32_16x16x32_bf16(false, af, false, w0, (short)0, a0, false, false);
        a1 = __builtin_amdgcn_wmma_f32_16x16x32_bf16(false, af, false, w1, (short)0, a1, false, false);
        a2 = __builtin_amdgcn_wmma_f32_16x16x32_bf16(false, af, false, w2, (short)0, a2, false, false);
        a3 = __builtin_amdgcn_wmma_f32_16x16x32_bf16(false, af, false, w3, (short)0, a3, false, false);
      }
    }
    // --- x half (K = 0..511) from the async-staged LDS tile ---
    wait_async0();
    __syncthreads();
    {
      const unsigned short* xbase = xs + (size_t)(wave * 16) * LDX_;
      for (int kk = 0; kk < H_; kk += 32) {
        bf16x16 af = load_frag(xbase, LDX_, lane, kk);
        bf16x16 w0 = load_frag(wlds + 0 * 16 * LDW_E, LDW_E, lane, kk);
        bf16x16 w1 = load_frag(wlds + 1 * 16 * LDW_E, LDW_E, lane, kk);
        bf16x16 w2 = load_frag(wlds + 2 * 16 * LDW_E, LDW_E, lane, kk);
        bf16x16 w3 = load_frag(wlds + 3 * 16 * LDW_E, LDW_E, lane, kk);
        a0 = __builtin_amdgcn_wmma_f32_16x16x32_bf16(false, af, false, w0, (short)0, a0, false, false);
        a1 = __builtin_amdgcn_wmma_f32_16x16x32_bf16(false, af, false, w1, (short)0, a1, false, false);
        a2 = __builtin_amdgcn_wmma_f32_16x16x32_bf16(false, af, false, w2, (short)0, a2, false, false);
        a3 = __builtin_amdgcn_wmma_f32_16x16x32_bf16(false, af, false, w3, (short)0, a3, false, false);
      }
    }
#pragma unroll
    for (int rr = 0; rr < 8; rr++) {
      float gi = a0[rr] + b_i, gf = a1[rr] + b_f;
      float gg = a2[rr] + b_g, go = a3[rr] + b_o;
      float cn = sigm(gf) * cst[rr] + sigm(gi) * tanh_(gg);
      float hh = sigm(go) * tanh_(cn);
      cst[rr] = cn;
      hnxt[(size_t)(rowbase + rr) * H_ + col] = f2bf(hh);
      if (t == T_ - 1) hfin[(size_t)(rowbase + rr) * H_ + col] = hh;
    }
    grid_bar(bar + 0, bar + 1, nb);
    unsigned short* tmp = hcur; hcur = hnxt; hnxt = tmp;
  }
}

// ---------------------------------------------------------------------------
// 5) Bottleneck head (tiny GEMMs, plain f32).
// ---------------------------------------------------------------------------
__global__ __launch_bounds__(256) void head1_kernel(
    const float* __restrict__ hf, const float* __restrict__ hb,
    const float* __restrict__ Wc, const float* __restrict__ bc,
    float* __restrict__ dout_z, float* __restrict__ dout_enc,
    float* __restrict__ z_ws) {
  int i = blockIdx.x * 256 + threadIdx.x;       // B*H
  int b = i >> 9, j = i & 511;
  dout_enc[i] = hf[i] + hb[i];
  if (j < CODE_) {
    const float* hfr = hf + (size_t)b * H_;
    const float* hbr = hb + (size_t)b * H_;
    const float* w = Wc + (size_t)j * H_;
    float s = bc[j];
    for (int k = 0; k < H_; k++) s += (hfr[k] + hbr[k]) * w[k];
    dout_z[b * CODE_ + j] = s;
    z_ws[b * CODE_ + j] = s;
  }
}

__global__ __launch_bounds__(256) void head2_kernel(
    const float* __restrict__ z, const float* __restrict__ Ws_,
    const float* __restrict__ bs_, const float* __restrict__ Wd,
    const float* __restrict__ bd, float* __restrict__ dout_static,
    unsigned short* __restrict__ dv_bf) {
  int i = blockIdx.x * 256 + threadIdx.x;       // B*H
  int b = i >> 9, j = i & 511;
  const float* zr = z + (size_t)b * CODE_;
  {
    const float* w = Wd + (size_t)j * CODE_;
    float s = bd[j];
    for (int k = 0; k < CODE_; k++) s += zr[k] * w[k];
    dv_bf[i] = f2bf(s);                         // decoder h_0
  }
  if (j < OUTC_) {
    const float* w = Ws_ + (size_t)j * CODE_;
    float s = bs_[j];
    for (int k = 0; k < CODE_; k++) s += zr[k] * w[k];
    dout_static[b * OUTC_ + j] = s;
  }
}

// ---------------------------------------------------------------------------
// 6) Decoder LSTM (persistent). Step 0: gates = dv @ Whh_d^T; t>=1:
//    gates = h @ (Wih_d + Whh_d)^T (xin == h). Both weight slices in LDS,
//    A-operand async-staged into LDS each step.
// ---------------------------------------------------------------------------
__global__ __launch_bounds__(256) void lstm_dec_kernel(
    const unsigned short* __restrict__ dv_bf, const unsigned short* __restrict__ Wdec0,
    const unsigned short* __restrict__ Wdec1, const float* __restrict__ bias,
    unsigned short* __restrict__ h_dec, unsigned* bar) {
  extern __shared__ unsigned short lds[];
  unsigned short* wlds = lds;                      // 8*16*LDX_ halves
  unsigned short* xs   = lds + 8 * 16 * LDX_;      // 128*LDX_ halves
  int lane = threadIdx.x & 31, wave = threadIdx.x >> 5;
  int n0 = blockIdx.x * 16;
  int mblock = blockIdx.y * 128;
  int m0 = mblock + wave * 16;

  { // async preload of both weight slices: 128 rows x 512 halves
    int row = threadIdx.x >> 1;                    // 0..127
    int part = threadIdx.x & 1;
    int s = row >> 6, grow = row & 63;
    int g = grow >> 4, rr = grow & 15;
    const unsigned short* W = s ? Wdec1 : Wdec0;
    const unsigned short* src = W + ((size_t)(g * H_ + n0 + rr)) * H_ + part * 256;
    unsigned short* dst = wlds + (size_t)row * LDX_ + part * 256;
#pragma unroll
    for (int i = 0; i < 256; i += 8)
      async_g2l_b128(dst + i, src + i);
  }
  wait_async0();
  __syncthreads();

  int col = n0 + (lane & 15);
  float b_i = bias[0 * H_ + col], b_f = bias[1 * H_ + col];
  float b_g = bias[2 * H_ + col], b_o = bias[3 * H_ + col];
  float cst[8];
#pragma unroll
  for (int rr = 0; rr < 8; rr++) cst[rr] = 0.f;
  const unsigned nb = gridDim.x * gridDim.y;
  int rowbase = m0 + ((lane & 16) ? 8 : 0);
  int srow = threadIdx.x >> 1;
  int spart = threadIdx.x & 1;

  for (int t = 0; t < T_; t++) {
    const unsigned short* hsrc =
        (t == 0) ? dv_bf : (h_dec + (size_t)(t - 1) * BH_);
    { // async stage of this step's A tile
      const unsigned short* src = hsrc + (size_t)(mblock + srow) * H_ + spart * 256;
      unsigned short* dst = xs + (size_t)srow * LDX_ + spart * 256;
#pragma unroll
      for (int i = 0; i < 256; i += 8)
        async_g2l_b128(dst + i, src + i);
    }
    wait_async0();
    __syncthreads();
    const unsigned short* wbase = wlds + (size_t)((t == 0) ? 0 : 4) * 16 * LDX_;
    const unsigned short* xbase = xs + (size_t)(wave * 16) * LDX_;
    f32x8 a0 = {}, a1 = {}, a2 = {}, a3 = {};
    for (int kk = 0; kk < H_; kk += 32) {
      bf16x16 af = load_frag(xbase, LDX_, lane, kk);
      bf16x16 w0 = load_frag(wbase + 0 * 16 * LDX_, LDX_, lane, kk);
      bf16x16 w1 = load_frag(wbase + 1 * 16 * LDX_, LDX_, lane, kk);
      bf16x16 w2 = load_frag(wbase + 2 * 16 * LDX_, LDX_, lane, kk);
      bf16x16 w3 = load_frag(wbase + 3 * 16 * LDX_, LDX_, lane, kk);
      a0 = __builtin_amdgcn_wmma_f32_16x16x32_bf16(false, af, false, w0, (short)0, a0, false, false);
      a1 = __builtin_amdgcn_wmma_f32_16x16x32_bf16(false, af, false, w1, (short)0, a1, false, false);
      a2 = __builtin_amdgcn_wmma_f32_16x16x32_bf16(false, af, false, w2, (short)0, a2, false, false);
      a3 = __builtin_amdgcn_wmma_f32_16x16x32_bf16(false, af, false, w3, (short)0, a3, false, false);
    }
    unsigned short* hout = h_dec + (size_t)t * BH_;
#pragma unroll
    for (int rr = 0; rr < 8; rr++) {
      float gi = a0[rr] + b_i, gf = a1[rr] + b_f;
      float gg = a2[rr] + b_g, go = a3[rr] + b_o;
      float cn = sigm(gf) * cst[rr] + sigm(gi) * tanh_(gg);
      float hh = sigm(go) * tanh_(cn);
      cst[rr] = cn;
      hout[(size_t)(rowbase + rr) * H_ + col] = f2bf(hh);
    }
    grid_bar(bar + 0, bar + 1, nb);
  }
}

// ---------------------------------------------------------------------------
// 7) Output projection: out[b][t][c] = h_dec[t][b][:] . Wo[c][:] + bo[c]
// ---------------------------------------------------------------------------
__global__ __launch_bounds__(256) void out_gemm_kernel(
    const unsigned short* __restrict__ h_dec, const unsigned short* __restrict__ Wo_bf,
    const float* __restrict__ bo, float* __restrict__ out) {
  int lane = threadIdx.x & 31, wave = threadIdx.x >> 5;
  int gw = blockIdx.x * 8 + wave;               // 0..23359
  int m0 = (gw >> 1) * 16;
  int n0 = (gw & 1) * 16;
  const unsigned short* abase = h_dec + (size_t)m0 * H_;
  const unsigned short* bbase = Wo_bf + (size_t)n0 * H_;
  f32x8 acc = {};
  for (int kk = 0; kk < H_; kk += 32) {
    bf16x16 af = load_frag(abase, H_, lane, kk);
    bf16x16 wf = load_frag(bbase, H_, lane, kk);
    acc = __builtin_amdgcn_wmma_f32_16x16x32_bf16(false, af, false, wf, (short)0, acc,
                                                  false, false);
  }
  int col = n0 + (lane & 15);
  float bv = bo[col];
  int rowbase = m0 + ((lane & 16) ? 8 : 0);
#pragma unroll
  for (int rr = 0; rr < 8; rr++) {
    int m = rowbase + rr, t = m >> 9, b = m & 511;
    out[((size_t)b * T_ + t) * CIN_ + col] = acc[rr] + bv;
  }
}

// ---------------------------------------------------------------------------
// Host side
// ---------------------------------------------------------------------------
extern "C" void kernel_launch(void* const* d_in, const int* in_sizes, int n_in,
                              void* d_out, int out_size, void* d_ws, size_t ws_size,
                              hipStream_t stream) {
  const float* x     = (const float*)d_in[0];
  const float* We    = (const float*)d_in[1];
  /* be = d_in[2]: unused — cancels exactly in BatchNorm */
  const float* gamma = (const float*)d_in[3];
  const float* beta  = (const float*)d_in[4];
  const float* Wih_f = (const float*)d_in[5];
  const float* Whh_f = (const float*)d_in[6];
  const float* b_f   = (const float*)d_in[7];
  const float* Wih_b = (const float*)d_in[8];
  const float* Whh_b = (const float*)d_in[9];
  const float* b_b   = (const float*)d_in[10];
  const float* Wc    = (const float*)d_in[11];
  const float* bc    = (const float*)d_in[12];
  const float* Ws_   = (const float*)d_in[13];
  const float* bs_   = (const float*)d_in[14];
  const float* Wd    = (const float*)d_in[15];
  const float* bd    = (const float*)d_in[16];
  const float* Wih_d = (const float*)d_in[17];
  const float* Whh_d = (const float*)d_in[18];
  const float* b_d   = (const float*)d_in[19];
  const float* Wo    = (const float*)d_in[20];
  const float* bo    = (const float*)d_in[21];

  // Workspace carve-up (requires ws_size >= ~403 MB)
  char* ws = (char*)d_ws;
  size_t off = 0;
  auto take = [&](size_t bytes) {
    char* p = ws + off;
    off += (bytes + 255) & ~(size_t)255;
    return p;
  };
  unsigned short* h_enc  = (unsigned short*)take((size_t)ROWS_ * H_ * 2);
  unsigned short* h_dec  = (unsigned short*)take((size_t)ROWS_ * H_ * 2);
  unsigned short* Wcat_f = (unsigned short*)take((size_t)G4_ * 1024 * 2);
  unsigned short* Wcat_b = (unsigned short*)take((size_t)G4_ * 1024 * 2);
  unsigned short* Wdec0  = (unsigned short*)take((size_t)G4_ * H_ * 2);
  unsigned short* Wdec1  = (unsigned short*)take((size_t)G4_ * H_ * 2);
  unsigned short* We_bf  = (unsigned short*)take((size_t)H_ * CIN_ * 2);
  unsigned short* Wo_bf  = (unsigned short*)take((size_t)CIN_ * H_ * 2);
  unsigned short* hp0    = (unsigned short*)take((size_t)BH_ * 2);
  unsigned short* hp1    = (unsigned short*)take((size_t)BH_ * 2);
  float*          hf_fin = (float*)take((size_t)BH_ * 4);
  float*          hb_fin = (float*)take((size_t)BH_ * 4);
  unsigned short* dv_bf  = (unsigned short*)take((size_t)BH_ * 2);
  float*          stats  = (float*)take((size_t)H_ * 2 * 4);
  float*          part   = (float*)take((size_t)NPART_ * H_ * 2 * 4);
  float*          z_ws   = (float*)take((size_t)B_ * CODE_ * 4);
  unsigned*       bar    = (unsigned*)take(256);

  float* dout = (float*)d_out;
  float* out_z      = dout;                                // 512*128
  float* out_enc    = dout + (size_t)B_ * CODE_;           // 512*512
  float* out_static = out_enc + (size_t)B_ * H_;           // 512*16
  float* out_seq    = out_static + (size_t)B_ * OUTC_;     // 512*365*32

  prep_kernel<<<1024, 256, 0, stream>>>(Wih_f, Whh_f, Wih_b, Whh_b, Wih_d, Whh_d,
                                        We, Wo, Wcat_f, Wcat_b, Wdec0, Wdec1,
                                        We_bf, Wo_bf, bar);

  enc_gemm_kernel<<<dim3(ROWS_ / 16, 4), 256, 0, stream>>>(x, We_bf, h_enc);
  bn_part_kernel<<<NPART_, 256, 0, stream>>>(h_enc, part);
  bn_finish_kernel<<<2, 256, 0, stream>>>(part, stats);
  bn_apply_kernel<<<(unsigned)(((size_t)ROWS_ * H_) / 256), 256, 0, stream>>>(
      h_enc, stats, gamma, beta);

  // LDS: weights (4*16*1032) + x stage (128*520), halves -> bytes
  const size_t lds_enc = ((size_t)4 * 16 * LDW_E + (size_t)128 * LDX_) * 2; // 265216
  lstm_enc_kernel<<<dim3(32, 4), 256, lds_enc, stream>>>(h_enc, Wcat_f, b_f, hp0, hp1,
                                                         hf_fin, bar, 0);
  lstm_enc_kernel<<<dim3(32, 4), 256, lds_enc, stream>>>(h_enc, Wcat_b, b_b, hp0, hp1,
                                                         hb_fin, bar, 1);

  head1_kernel<<<BH_ / 256, 256, 0, stream>>>(hf_fin, hb_fin, Wc, bc, out_z, out_enc,
                                              z_ws);
  head2_kernel<<<BH_ / 256, 256, 0, stream>>>(z_ws, Ws_, bs_, Wd, bd, out_static,
                                              dv_bf);

  // LDS: weights (8*16*520) + A stage (128*520), halves -> bytes
  const size_t lds_dec = ((size_t)8 * 16 * LDX_ + (size_t)128 * LDX_) * 2;  // 266240
  lstm_dec_kernel<<<dim3(32, 4), 256, lds_dec, stream>>>(dv_bf, Wdec0, Wdec1, b_d,
                                                         h_dec, bar);

  out_gemm_kernel<<<(ROWS_ / 16) * 2 / 8, 256, 0, stream>>>(h_dec, Wo_bf, bo, out_seq);
}
